// GRU_36575941492856
// MI455X (gfx1250) — matrix operand order repaired and checked by reading, not compiled, not run on hardware
//
#include <hip/hip_runtime.h>
#include <hip/hip_bf16.h>
#include <stdint.h>

// ---------------------------------------------------------------------------
// GRU on MI455X (gfx1250, wave32, WMMA + async global->LDS).
//
//   Kernel 1 (pack_weights): repack Uz|Ur|U into bf16 WMMA B-operand tiles.
//   Kernel 2 (input_proj): P[t,b,0:768] = x@[Wz|Wr|W] + bias, bf16 WMMA GEMM,
//     result stored bf16 in workspace (96 MB).
//   Kernel 3 (gru_recurrent): 4 persistent workgroups, each owns 16 batch
//     rows (one WMMA M-tile), full 1024-step scan. U weights live in VGPRs,
//     h/z/r in LDS. The per-step P slice (16x768 bf16 = 24 KB, contiguous)
//     is double-buffered in LDS via GLOBAL_LOAD_ASYNC_TO_LDS_B128 issued one
//     step ahead (ASYNCcnt-tracked), so no global-memory latency sits on the
//     serial recurrence.
//
// Workspace layout (requires ws_size >= 512KB + 96MB):
//   [0,       256KB) : 256 zr-weight tiles (bf16)
//   [256KB,   384KB) : 128 U-weight tiles  (bf16)
//   [512KB, 512KB+65536*768*2) : P (bf16)
// ---------------------------------------------------------------------------

typedef __attribute__((ext_vector_type(16))) __bf16 v16bf;
typedef __attribute__((ext_vector_type(8)))  float  v8f;

#define WS_P_OFF (512 * 1024)

union Frag { v16bf v; uint4 q[2]; };

__device__ __forceinline__ __bf16 f2bf(float f) {
  union { float f; uint32_t u; } a; a.f = f;
  uint32_t r = a.u + 0x7FFFu + ((a.u >> 16) & 1u);   // round-to-nearest-even
  union { unsigned short s; __bf16 b; } o; o.s = (unsigned short)(r >> 16);
  return o.b;
}

__device__ __forceinline__ v8f vzero() {
  v8f z = {0.f, 0.f, 0.f, 0.f, 0.f, 0.f, 0.f, 0.f};
  return z;
}

__device__ __forceinline__ v8f wmma_bf16(const Frag& A, const Frag& B, v8f C) {
  return __builtin_amdgcn_wmma_f32_16x16x32_bf16(false, A.v, false, B.v,
                                                 (short)0, C, false, false);
}

__device__ __forceinline__ float sigmoidf_(float x) {
  return 1.0f / (1.0f + __expf(-x));
}

// Async copy 16 B from global to LDS (per lane). Tracked by ASYNCcnt.
__device__ __forceinline__ void async_b128(uint32_t lds_off, const void* g) {
  asm volatile("global_load_async_to_lds_b128 %0, %1, off"
               :: "v"(lds_off), "v"(g)
               : "memory");
}

// A fragment (16x32 bf16): row m = lane&15 ; lanes 0-15 hold K=0..15,
// lanes 16-31 K=16..31 (matches B operand order).
__device__ __forceinline__ Frag load_fragA(const __bf16* buf, int stride,
                                           int ks, int l) {
  Frag f;
  const uint4* p = (const uint4*)(buf + (l & 15) * stride + ks * 32);
  int hi = l >> 4;
  f.q[0] = p[2 * hi];
  f.q[1] = p[2 * hi + 1];
  return f;
}

// B fragment (32x16 bf16) from an LDS tile stored transposed Bt[n][32]:
// col n = lane&15 ; lanes 0-15 hold K=0..15, lanes 16-31 K=16..31.
__device__ __forceinline__ Frag load_fragB(const __bf16* buf, int stride,
                                           int ks, int l) {
  Frag f;
  const uint4* p =
      (const uint4*)(buf + (l & 15) * stride + ks * 32 + 16 * (l >> 4));
  f.q[0] = p[0];
  f.q[1] = p[1];
  return f;
}

// ---------------------------------------------------------------------------
// Kernel 1: pack U weights into per-lane WMMA B-operand tiles.
// grid = 384 (tiles), block = 32. Tile t<256: zr space (Uz|Ur); t>=256: U.
// ---------------------------------------------------------------------------
__global__ void pack_weights(const float* __restrict__ Uz,
                             const float* __restrict__ Ur,
                             const float* __restrict__ U,
                             __bf16* __restrict__ wtiles) {
  int tile = blockIdx.x;
  int l = threadIdx.x;
  const float* src;
  int col, ks;
  if (tile < 256) {
    int nt = tile >> 3; ks = tile & 7;
    int cg = nt * 16 + (l & 15);
    src = (cg < 256) ? Uz : Ur;
    col = cg & 255;
  } else {
    int tt = tile - 256;
    int nt = tt >> 3; ks = tt & 7;
    col = nt * 16 + (l & 15);
    src = U;
  }
  __bf16* out = wtiles + tile * 512 + l * 16;
  int k0 = ks * 32 + 16 * (l >> 4);
#pragma unroll
  for (int e = 0; e < 16; ++e)
    out[e] = f2bf(src[(k0 + e) * 256 + col]);
}

// ---------------------------------------------------------------------------
// Kernel 2: P = bf16(X @ [Wz|Wr|W] + bias).  grid = 1024*6, block = 256.
// WG tile: 64 rows x 128 cols. wave w: M-subtile w>>1, N-subtiles (w&1)*4+[0,4).
// ---------------------------------------------------------------------------
__global__ void __launch_bounds__(256)
input_proj(const float* __restrict__ X,  const float* __restrict__ Wz,
           const float* __restrict__ Wr, const float* __restrict__ W,
           const float* __restrict__ bz, const float* __restrict__ br,
           const float* __restrict__ b,  __bf16* __restrict__ P) {
  __shared__ __align__(16) __bf16 Abf[64][32];
  __shared__ __align__(16) __bf16 Btf[128][32];

  int mb = blockIdx.x & 1023;
  int nb = blockIdx.x >> 10;
  int tid = threadIdx.x, w = tid >> 5, l = tid & 31;
  int rows0 = mb * 64, c0 = nb * 128;
  int mt = w >> 1, ntg = (w & 1) * 4;

  v8f acc[4];
#pragma unroll
  for (int j = 0; j < 4; ++j) acc[j] = vzero();

  for (int ks = 0; ks < 8; ++ks) {
#pragma unroll
    for (int j = 0; j < 8; ++j) {
      int i = tid * 8 + j;
      int m = i >> 5, kk = i & 31;
      Abf[m][kk] = f2bf(X[(size_t)(rows0 + m) * 256 + ks * 32 + kk]);
    }
#pragma unroll
    for (int j = 0; j < 16; ++j) {
      int i = tid * 16 + j;
      int n = i >> 5, kk = i & 31;
      int c = c0 + n, k = ks * 32 + kk;
      const float* src = (c < 256) ? Wz : (c < 512) ? Wr : W;
      Btf[n][kk] = f2bf(src[k * 256 + (c & 255)]);
    }
    __syncthreads();

    Frag A = load_fragA(&Abf[mt * 16][0], 32, 0, l);
#pragma unroll
    for (int j = 0; j < 4; ++j) {
      Frag B = load_fragB(&Btf[(ntg + j) * 16][0], 32, 0, l);
      acc[j] = wmma_bf16(A, B, acc[j]);
    }
    __syncthreads();
  }

  int hi = l >> 4, n = l & 15;
#pragma unroll
  for (int j = 0; j < 4; ++j) {
    int c = c0 + (ntg + j) * 16 + n;
    const float* bsrc = (c < 256) ? bz : (c < 512) ? br : b;
    float bias = bsrc[c & 255];
#pragma unroll
    for (int v = 0; v < 8; ++v) {
      int row = rows0 + mt * 16 + v + 8 * hi;
      P[(size_t)row * 768 + c] = f2bf(acc[j][v] + bias);
    }
  }
}

// ---------------------------------------------------------------------------
// Kernel 3: persistent recurrent scan.  grid = 4, block = 512 (16 waves).
// WG owns batch rows [16*wg, 16*wg+16). Weights in VGPRs, state in LDS,
// P slices double-buffered in LDS via async global->LDS prefetch.
// ---------------------------------------------------------------------------
__global__ void __launch_bounds__(512)
gru_recurrent(const __bf16* __restrict__ wtiles, const __bf16* __restrict__ P,
              const float* __restrict__ h0, float* __restrict__ out) {
  __shared__ __align__(16) float  h_s[16][256];     // 16 KB
  __shared__ __align__(16) float  zr_s[16][512];    // 32 KB
  __shared__ __align__(16) __bf16 h_bf[16][256];    //  8 KB
  __shared__ __align__(16) __bf16 rh_bf[16][256];   //  8 KB
  __shared__ __align__(16) __bf16 Pl[2][16][768];   // 48 KB double buffer

  int wg = blockIdx.x;                   // 0..3
  int tid = threadIdx.x, w = tid >> 5, l = tid & 31;
  const int prow0 = wg * 16;

  // ---- preload weight fragments into VGPRs (persistent across T loop) ----
  Frag uzr[2][8];                        // phase1: N-tiles {2w, 2w+1} of 32
  Frag uu[8];                            // phase2: N-tile w of 16
#pragma unroll
  for (int a = 0; a < 2; ++a)
#pragma unroll
    for (int ks = 0; ks < 8; ++ks) {
      const uint4* p =
          (const uint4*)(wtiles + ((2 * w + a) * 8 + ks) * 512 + l * 16);
      uzr[a][ks].q[0] = p[0];
      uzr[a][ks].q[1] = p[1];
    }
#pragma unroll
  for (int ks = 0; ks < 8; ++ks) {
    const uint4* p =
        (const uint4*)(wtiles + 256 * 512 + (w * 8 + ks) * 512 + l * 16);
    uu[ks].q[0] = p[0];
    uu[ks].q[1] = p[1];
  }

  // LDS byte offsets of the two P buffers (flat shared addr low 32 bits).
  uint32_t plofs[2];
  plofs[0] = (uint32_t)(uintptr_t)&Pl[0][0][0];
  plofs[1] = (uint32_t)(uintptr_t)&Pl[1][0][0];

  // ---- prologue: async-prefetch P slice for t=0 into buffer 0 ----
  {
    const __bf16* g = P + (size_t)prow0 * 768 + tid * 24;   // 48 B / thread
    uint32_t d = plofs[0] + tid * 48;
#pragma unroll
    for (int j = 0; j < 3; ++j) async_b128(d + j * 16, g + j * 8);
  }

  // ---- init h = broadcast(init_hidden) ----
#pragma unroll
  for (int j = 0; j < 8; ++j) {
    int i = tid * 8 + j;                 // 4096 elements
    h_s[i >> 8][i & 255] = h0[i & 255];
  }
  __syncthreads();

  const int hi = l >> 4, n = l & 15;

  for (int t = 0; t < 1024; ++t) {
    const int cur = t & 1, nxt = cur ^ 1;

    // issue async prefetch of next step's P slice (wraps harmlessly at end)
    {
      int tp = (t + 1) & 1023;
      const __bf16* g = P + ((size_t)tp * 64 + prow0) * 768 + tid * 24;
      uint32_t d = plofs[nxt] + tid * 48;
#pragma unroll
      for (int j = 0; j < 3; ++j) async_b128(d + j * 16, g + j * 8);
    }

    // emit pre-update hidden: out[t][prow0..+16][*] = h_s  (16 KB, f32)
    float* outp = out + ((size_t)t * 64 + prow0) * 256;
    const float* hs = &h_s[0][0];
#pragma unroll
    for (int j = 0; j < 2; ++j) {
      int i = tid + j * 512;             // 1024 float4s
      ((float4*)outp)[i] = ((const float4*)hs)[i];
    }
    // bf16 copy of h for the A operand
#pragma unroll
    for (int j = 0; j < 8; ++j) {
      int i = tid * 8 + j;
      h_bf[i >> 8][i & 255] = f2bf(hs[i]);
    }

    // wait for THIS step's P slice (3 newest outstanding = next step's)
    asm volatile("s_wait_asynccnt 0x3" ::: "memory");
    __syncthreads();

    // ---- phase 1: [16x256] @ [256x512] -> z|r pre-activations ----
    v8f a0 = vzero(), a1 = vzero();
#pragma unroll
    for (int ks = 0; ks < 8; ++ks) {
      Frag A = load_fragA(&h_bf[0][0], 256, ks, l);
      a0 = wmma_bf16(A, uzr[0][ks], a0);
      a1 = wmma_bf16(A, uzr[1][ks], a1);
    }
#pragma unroll
    for (int a = 0; a < 2; ++a) {
      int c = 32 * w + 16 * a + n;       // zr column (0..511)
      v8f accv = a ? a1 : a0;
#pragma unroll
      for (int v = 0; v < 8; ++v) {
        int row = v + 8 * hi;
        float pv = (float)Pl[cur][row][c];
        zr_s[row][c] = sigmoidf_(accv[v] + pv);
      }
    }
    __syncthreads();

    // ---- r .* h -> bf16 A operand for phase 2 ----
#pragma unroll
    for (int j = 0; j < 8; ++j) {
      int i = tid * 8 + j;
      int m = i >> 8, c = i & 255;
      rh_bf[m][c] = f2bf(zr_s[m][256 + c] * h_s[m][c]);
    }
    __syncthreads();

    // ---- phase 2: [16x256] @ [256x256] -> h_bar; update h ----
    v8f a2 = vzero();
#pragma unroll
    for (int ks = 0; ks < 8; ++ks) {
      Frag A = load_fragA(&rh_bf[0][0], 256, ks, l);
      a2 = wmma_bf16(A, uu[ks], a2);
    }
    {
      int c = 16 * w + n;                // hidden column (0..255)
#pragma unroll
      for (int v = 0; v < 8; ++v) {
        int row = v + 8 * hi;
        float pv = (float)Pl[cur][row][512 + c];
        float hb = tanhf(a2[v] + pv);
        float r = zr_s[row][256 + c];
        float z = zr_s[row][c];
        float h = h_s[row][c];
        h_s[row][c] = (1.0f - r) * h + z * hb;   // nonstandard (1-r) per ref
      }
    }
    __syncthreads();
  }
}

// ---------------------------------------------------------------------------
extern "C" void kernel_launch(void* const* d_in, const int* in_sizes, int n_in,
                              void* d_out, int out_size, void* d_ws,
                              size_t ws_size, hipStream_t stream) {
  const float* X  = (const float*)d_in[0];   // [1024,64,256]
  const float* W  = (const float*)d_in[1];   // h_bar input weight
  const float* U  = (const float*)d_in[2];   // h_bar recurrent weight
  const float* b  = (const float*)d_in[3];
  const float* Wr = (const float*)d_in[4];
  const float* Ur = (const float*)d_in[5];
  const float* br = (const float*)d_in[6];
  const float* Wz = (const float*)d_in[7];
  const float* Uz = (const float*)d_in[8];
  const float* bz = (const float*)d_in[9];
  const float* h0 = (const float*)d_in[10];  // [1,256]

  __bf16* wtiles = (__bf16*)d_ws;
  __bf16* P      = (__bf16*)((char*)d_ws + WS_P_OFF);

  pack_weights<<<384, 32, 0, stream>>>(Uz, Ur, U, wtiles);
  input_proj<<<1024 * 6, 256, 0, stream>>>(X, Wz, Wr, W, bz, br, b, P);
  gru_recurrent<<<4, 512, 0, stream>>>(wtiles, P, h0, (float*)d_out);
}
